// SGConv3_61581241090461
// MI455X (gfx1250) — matrix operand order repaired and checked by reading, not compiled, not run on hardware
//
#include <hip/hip_runtime.h>
#include <hip/hip_bf16.h>
#include <math.h>

// ---------------------------------------------------------------------------
// SGConv3 forward for MI455X (gfx1250, wave32, WMMA).
// All heavy math on v_wmma_f32_16x16x32_bf16 (bf16 in, f32 accumulate).
// Causal long convs are lower-triangular Toeplitz x tall-skinny WMMA GEMMs.
// B operands are staged in LDS in K-contiguous (transposed) layout so every
// fragment is two aligned ds_load_b128; A operands are two global_load_b128.
// ---------------------------------------------------------------------------

#define B_   4
#define DM   1024
#define LL   2048
#define HD   4
#define HH   256
#define KD_  64
#define NS_  6
#define BSP  40        // padded LDS row stride (bf16) for 32-K tiles
#define SIGP (LL + 8)  // padded LDS row stride for signal rows

typedef __attribute__((ext_vector_type(16))) __bf16 v16bf;
typedef __attribute__((ext_vector_type(8)))  __bf16 v8bf;
typedef __attribute__((ext_vector_type(2)))  __bf16 v2bf;
typedef __attribute__((ext_vector_type(8)))  float  v8f;

// 16-bit A-matrix 16x32 layout (ISA 7.12.2): lanes 0-15 row M=lane hold
// K = {0..7, 16..23}; lanes 16-31 hold K = {8..15, 24..31}.
static __device__ __forceinline__ int kmapA(int e, int lane) {
    return ((e < 8) ? e : (e + 8)) + ((lane >> 4) << 3);
}

static __device__ __forceinline__ v8f wmma_bf16(v16bf a, v16bf b, v8f c) {
    return __builtin_amdgcn_wmma_f32_16x16x32_bf16(false, a, false, b,
                                                   (short)0, c, false, false);
}

// B fragment: lane&15 = N column; element e => K = kmapA(e, lane).
// With a K-contiguous row pointer this is two aligned 16B LDS vector loads.
static __device__ __forceinline__ v16bf load_bfrag(const __bf16* brow, int lane) {
    const __bf16* p = brow + ((lane >> 4) << 3);
    v8bf b0 = *(const v8bf*)(p);
    v8bf b1 = *(const v8bf*)(p + 16);
    v16bf b;
#pragma unroll
    for (int e = 0; e < 8; ++e) { b[e] = b0[e]; b[e + 8] = b1[e]; }
    return b;
}

static __device__ __forceinline__ float gelu_exact(float x) {
    return 0.5f * x * (1.f + erff(x * 0.70710678118654752f));
}

// ---------------------------------------------------------------------------
// Kernel 0: one-time f32 -> bf16 conversion (weights + u).
// ---------------------------------------------------------------------------
__global__ void f32_to_bf16_kernel(const float* __restrict__ src,
                                   __bf16* __restrict__ dst, int n) {
    int i = (int)(blockIdx.x * blockDim.x + threadIdx.x);
    int stride = (int)(gridDim.x * blockDim.x);
    for (; i < n; i += stride) dst[i] = (__bf16)src[i];
}

// ---------------------------------------------------------------------------
// Kernel 1: build normalized multi-scale kernels k_key (1024 x L) and
// k_v (256 x L).  One 256-thread block per output channel.
// ---------------------------------------------------------------------------
__global__ void build_kernels_kernel(const float* __restrict__ kkey_src,
                                     const float* __restrict__ kv_src,
                                     float* __restrict__ k_key,
                                     float* __restrict__ k_v) {
    __shared__ float red[256];
    int ch = blockIdx.x;
    const float* src;
    float* dst;
    int hdim;
    if (ch < DM) { src = kkey_src; dst = k_key; hdim = DM; }
    else         { ch -= DM; src = kv_src; dst = k_v; hdim = HH; }

    float vals[8];
    float ss = 0.f;
#pragma unroll
    for (int t = 0; t < 8; ++t) {
        int l = (int)threadIdx.x + t * 256;
        int i, off, scale;
        if (l < 64) { i = 0; off = 0; scale = 1; }
        else {
            i = 32 - __clz((unsigned)(l >> 6));   // 1..5
            off = 64 << (i - 1);
            scale = 1 << (i - 1);
        }
        int j = l - off;
        float pos = (j + 0.5f) / (float)scale - 0.5f;
        pos = fminf(fmaxf(pos, 0.f), 63.f);
        int lo = (int)floorf(pos);
        int hi = min(lo + 1, KD_ - 1);
        float w = pos - (float)lo;
        const float* x = src + ((size_t)i * hdim + ch) * KD_;
        float decay = (float)(1 << (5 - i));
        float v = (x[lo] * (1.f - w) + x[hi] * w) * decay;
        vals[t] = v;
        ss += v * v;
    }
    red[threadIdx.x] = ss;
    __syncthreads();
    for (int st = 128; st > 0; st >>= 1) {
        if ((int)threadIdx.x < st) red[threadIdx.x] += red[threadIdx.x + st];
        __syncthreads();
    }
    float inv = 1.f / sqrtf(red[0]);
#pragma unroll
    for (int t = 0; t < 8; ++t)
        dst[(size_t)ch * LL + threadIdx.x + t * 256] = vals[t] * inv;
}

// ---------------------------------------------------------------------------
// Kernel 2: Y[b,o,l] = sum_d W[o,d] * X[b,d,l] + bias[o]  (bf16 WMMA, f32 acc)
// Block = 256 threads (8 waves) covering a 128x16 output tile; the 32x16 B
// tile per K-step is staged TRANSPOSED in LDS once and shared by 8 waves, so
// each B fragment is two aligned ds_load_b128.  A = two global_load_b128.
// grid = (512 col tiles, 8 row groups), block = 256.
// ---------------------------------------------------------------------------
__global__ void gemm1024_kernel(const __bf16* __restrict__ Wbf,
                                const float* __restrict__ bias,
                                const __bf16* __restrict__ Xbf,
                                __bf16* __restrict__ Ybf,
                                float* __restrict__ Yf) {
    __shared__ __bf16 bsT[16][BSP];    // [n][k], padded rows
    const int tid  = (int)threadIdx.x;
    const int lane = tid & 31;
    const int wv   = tid >> 5;
    const int t  = (int)blockIdx.x;
    const int b  = t >> 7;             // 128 col tiles per batch
    const int l0 = (t & 127) << 4;
    const int o0 = ((int)blockIdx.y << 7) + (wv << 4);
    const size_t DL = (size_t)DM * LL;
    const __bf16* Xb = Xbf + (size_t)b * DL + l0;

    v8f acc = {0.f, 0.f, 0.f, 0.f, 0.f, 0.f, 0.f, 0.f};
    const int rowA = lane & 15;
    const int g    = (lane >> 4) << 3;
    const __bf16* Wrow = Wbf + (size_t)(o0 + rowA) * DM;
    const int skk = (tid * 2) >> 4;    // staging: k row 0..31
    const int snn = (tid * 2) & 15;    // staging: 2 consecutive n

    for (int k0 = 0; k0 < DM; k0 += 32) {
        __syncthreads();
        // stage B tile transposed: bsT[n][k] = X[k0+k][l0+n]
        v2bf xv = *(const v2bf*)(Xb + (size_t)(k0 + skk) * LL + snn);
        bsT[snn][skk]     = xv[0];
        bsT[snn + 1][skk] = xv[1];
        __syncthreads();

        v8bf a0 = *(const v8bf*)(Wrow + k0 + g);
        v8bf a1 = *(const v8bf*)(Wrow + k0 + g + 16);
        v16bf a;
#pragma unroll
        for (int e = 0; e < 8; ++e) { a[e] = a0[e]; a[e + 8] = a1[e]; }
        v16bf bb = load_bfrag(&bsT[lane & 15][0], lane);

        acc = wmma_bf16(a, bb, acc);
    }
    // C/D layout: lane&15 = N (=l), VGPR r -> M = r + 8*(lane>=16)
#pragma unroll
    for (int r = 0; r < 8; ++r) {
        int o = o0 + r + ((lane >> 4) << 3);
        int l = l0 + (lane & 15);
        float val = acc[r] + bias[o];
        size_t idx = (size_t)b * DL + (size_t)o * LL + l;
        if (Yf) Yf[idx] = val;
        else    Ybf[idx] = (__bf16)val;
    }
}

// ---------------------------------------------------------------------------
// Kernel 3: key-side causal conv + D_key skip (bf16 in/out).
// out[b,c,l] = sum_{m<=l} k_key[c,l-m]*kk[b,c,m] + D_key[c]*kk[b,c,l]
// Toeplitz(16x32) x signal(32x16); N columns = 4 batches + 12 zero-padded
// via a zeroed LDS row, so B fragments are two aligned ds_load_b128.
// grid = (L/128, 1024 channels), block = 256 (8 waves share LDS staging).
// ---------------------------------------------------------------------------
__global__ void key_conv_kernel(const float* __restrict__ kconv,   // (DM, LL) f32
                                const __bf16* __restrict__ kk,     // (B, DM, LL)
                                const float* __restrict__ Dkey,    // (DM)
                                __bf16* __restrict__ out) {        // (B, DM, LL)
    __shared__ __bf16 kern_s[LL];           // 4 KB
    __shared__ __bf16 sig_s[B_][SIGP];      // 16 KB (padded rows)
    __shared__ __bf16 zrow_s[SIGP];         // 4 KB of zeros
    const int c = (int)blockIdx.y;
    const size_t DL = (size_t)DM * LL;
    const __bf16 bz = (__bf16)0.f;

    for (int i = (int)threadIdx.x; i < LL; i += 256) {
        kern_s[i] = (__bf16)kconv[(size_t)c * LL + i];
        zrow_s[i] = bz;
#pragma unroll
        for (int bb = 0; bb < B_; ++bb)
            sig_s[bb][i] = kk[(size_t)bb * DL + (size_t)c * LL + i];
    }
    __syncthreads();

    const int wave = (int)threadIdx.x >> 5;
    const int lane = (int)threadIdx.x & 31;
    const int l0 = ((int)blockIdx.x << 7) + (wave << 4);
    const int rowA = lane & 15;
    const int nB = lane & 15;
    const __bf16* brow = (nB < B_) ? &sig_s[nB][0] : &zrow_s[0];

    v8f acc = {0.f, 0.f, 0.f, 0.f, 0.f, 0.f, 0.f, 0.f};
    for (int m0 = 0; m0 <= l0 + 15; m0 += 32) {     // m0 <= 2016, m <= 2047
        v16bf a;
#pragma unroll
        for (int e = 0; e < 16; ++e) {
            int idx = l0 + rowA - (m0 + kmapA(e, lane));
            a[e] = (idx >= 0) ? kern_s[max(idx, 0)] : bz;
        }
        v16bf bb = load_bfrag(brow + m0, lane);
        acc = wmma_bf16(a, bb, acc);
    }

    float dk = Dkey[c];
#pragma unroll
    for (int r = 0; r < 8; ++r) {
        int l = l0 + r + ((lane >> 4) << 3);
        if (nB < B_) {
            float skip = (float)sig_s[nB][l] * dk;
            out[(size_t)nB * DL + (size_t)c * LL + l] = (__bf16)(acc[r] + skip);
        }
    }
}

// ---------------------------------------------------------------------------
// Kernel 4: fused kv outer product + value-side causal conv + D skip +
// q gating (sum over d1 via wave32 __shfl_xor) + exact GELU.
// Per K-step the 32x16 kv product tile is staged TRANSPOSED in LDS once per
// block (shared by 8 waves); the K loop is block-uniform -- waves past their
// causal range execute WMMAs on all-zero Toeplitz A fragments (harmless).
// Output channel layout matches reshape(B, d2, h, L) -> (B, d2*256+h, L).
// grid = (L/128, 256 heads, 4 batch), block = 256.
// ---------------------------------------------------------------------------
__global__ void kv_conv_gate_kernel(const __bf16* __restrict__ kkc,   // (B, DM, LL)
                                    const __bf16* __restrict__ vv,    // (B, DM, LL)
                                    const __bf16* __restrict__ qq,    // (B, DM, LL)
                                    const float* __restrict__ kvkern, // (HH, LL) f32
                                    const float* __restrict__ Dv,     // (HH)
                                    __bf16* __restrict__ ygate) {     // (B, DM, LL)
    __shared__ __bf16 kern_s[LL];          // 4 KB
    __shared__ __bf16 kk_s[HD][SIGP];      // 16 KB
    __shared__ __bf16 v_s [HD][SIGP];      // 16 KB
    __shared__ __bf16 kvT[16][BSP];        // 1.25 KB  [n][k]
    const int h = (int)blockIdx.y;
    const int b = (int)blockIdx.z;
    const size_t DL = (size_t)DM * LL;
    const __bf16 bz = (__bf16)0.f;

    for (int i = (int)threadIdx.x; i < LL; i += 256) {
        kern_s[i] = (__bf16)kvkern[(size_t)h * LL + i];
#pragma unroll
        for (int d = 0; d < HD; ++d) {
            size_t base = (size_t)b * DL + (size_t)(h * HD + d) * LL + i;
            kk_s[d][i] = kkc[base];
            v_s [d][i] = vv[base];
        }
    }
    __syncthreads();

    const int tid  = (int)threadIdx.x;
    const int wave = tid >> 5;
    const int lane = tid & 31;
    const int l0 = ((int)blockIdx.x << 7) + (wave << 4);
    const int n  = lane & 15;
    const int d1 = n >> 2;
    const int d2 = n & 3;
    const int smm = (tid * 2) >> 4;        // staging: k row 0..31
    const int snn = (tid * 2) & 15;        // staging: 2 consecutive n (same d1)
    const int sd1 = snn >> 2;
    const int lmax = ((int)blockIdx.x << 7) + 127;   // block-uniform bound

    v8f acc = {0.f, 0.f, 0.f, 0.f, 0.f, 0.f, 0.f, 0.f};
    for (int m0 = 0; m0 <= lmax; m0 += 32) {         // m0 <= 2016, m <= 2047
        __syncthreads();
        {   // stage kv tile transposed: kvT[n][k] = kk[d1][m0+k]*v[d2][m0+k]
            int mm = m0 + smm;
            float kkv = (float)kk_s[sd1][mm];
            kvT[snn][smm]     = (__bf16)(kkv * (float)v_s[snn & 3][mm]);
            kvT[snn + 1][smm] = (__bf16)(kkv * (float)v_s[(snn + 1) & 3][mm]);
        }
        __syncthreads();

        v16bf a;
#pragma unroll
        for (int e = 0; e < 16; ++e) {
            int idx = l0 + (lane & 15) - (m0 + kmapA(e, lane));
            a[e] = (idx >= 0) ? kern_s[max(idx, 0)] : bz;
        }
        v16bf bb = load_bfrag(&kvT[n][0], lane);
        acc = wmma_bf16(a, bb, acc);
    }

    const float Dh = Dv[h];
#pragma unroll
    for (int r = 0; r < 8; ++r) {
        int l = l0 + r + ((lane >> 4) << 3);
        float kvl = (float)kk_s[d1][l] * (float)v_s[d2][l];
        float val = acc[r] + Dh * kvl;
        float qv = (float)qq[(size_t)b * DL + (size_t)(h * HD + d1) * LL + l];
        float p = val * qv;
        p += __shfl_xor(p, 4, 32);   // reduce over d1 bit 0
        p += __shfl_xor(p, 8, 32);   // reduce over d1 bit 1
        if (d1 == 0)
            ygate[(size_t)b * DL + (size_t)(d2 * HH + h) * LL + l] =
                (__bf16)gelu_exact(p);
    }
}

// ---------------------------------------------------------------------------
// Host-side orchestration (graph-capture safe: only async launches on stream).
// ---------------------------------------------------------------------------
extern "C" void kernel_launch(void* const* d_in, const int* in_sizes, int n_in,
                              void* d_out, int out_size, void* d_ws, size_t ws_size,
                              hipStream_t stream) {
    (void)in_sizes; (void)n_in; (void)out_size; (void)ws_size;

    const float* u     = (const float*)d_in[0];
    const float* q_w   = (const float*)d_in[1];
    const float* q_b   = (const float*)d_in[2];
    const float* k_w   = (const float*)d_in[3];
    const float* k_b   = (const float*)d_in[4];
    const float* v_w   = (const float*)d_in[5];
    const float* v_b   = (const float*)d_in[6];
    const float* kkeys = (const float*)d_in[7];   // (NS, 1024, 64)
    const float* kvs   = (const float*)d_in[8];   // (NS, 256, 64)
    const float* D_key = (const float*)d_in[9];   // (1, 1024)
    const float* Dv    = (const float*)d_in[10];  // (1, 256)
    const float* pw_w  = (const float*)d_in[11];
    const float* pw_b  = (const float*)d_in[12];

    const size_t DL   = (size_t)DM * LL;        // 2,097,152
    const size_t NBDL = (size_t)B_ * DL;        // 8,388,608
    const size_t NW   = (size_t)DM * DM;        // 1,048,576

    char* p = (char*)d_ws;
    float*  k_key = (float*)p;  p += (size_t)DM * LL * 4;   //  8 MB
    float*  k_v   = (float*)p;  p += (size_t)HH * LL * 4;   //  2 MB
    __bf16* u_bf  = (__bf16*)p; p += NBDL * 2;              // 16 MB
    __bf16* wq_bf = (__bf16*)p; p += NW * 2;                //  2 MB
    __bf16* wk_bf = (__bf16*)p; p += NW * 2;
    __bf16* wv_bf = (__bf16*)p; p += NW * 2;
    __bf16* wp_bf = (__bf16*)p; p += NW * 2;
    __bf16* qb    = (__bf16*)p; p += NBDL * 2;              // 16 MB each
    __bf16* kkp   = (__bf16*)p; p += NBDL * 2;
    __bf16* vb    = (__bf16*)p; p += NBDL * 2;
    __bf16* kkc   = (__bf16*)p; p += NBDL * 2;
    __bf16* yg    = (__bf16*)p; p += NBDL * 2;

    // 0. one-time bf16 conversion of u and the four weight matrices
    f32_to_bf16_kernel<<<2048, 256, 0, stream>>>(u, u_bf, (int)NBDL);
    f32_to_bf16_kernel<<<1024, 256, 0, stream>>>(q_w, wq_bf, (int)NW);
    f32_to_bf16_kernel<<<1024, 256, 0, stream>>>(k_w, wk_bf, (int)NW);
    f32_to_bf16_kernel<<<1024, 256, 0, stream>>>(v_w, wv_bf, (int)NW);
    f32_to_bf16_kernel<<<1024, 256, 0, stream>>>(pw_w, wp_bf, (int)NW);

    // 1. multi-scale kernel construction + normalization
    build_kernels_kernel<<<DM + HH, 256, 0, stream>>>(kkeys, kvs, k_key, k_v);

    // 2. q / k / v projections (bf16 WMMA, shared transposed-B LDS tiles)
    gemm1024_kernel<<<dim3(512, 8), 256, 0, stream>>>(wq_bf, q_b, u_bf, qb, nullptr);
    gemm1024_kernel<<<dim3(512, 8), 256, 0, stream>>>(wk_bf, k_b, u_bf, kkp, nullptr);
    gemm1024_kernel<<<dim3(512, 8), 256, 0, stream>>>(wv_bf, v_b, u_bf, vb, nullptr);

    // 3. key-side causal Toeplitz conv + D_key skip
    key_conv_kernel<<<dim3(LL / 128, DM), 256, 0, stream>>>(k_key, kkp, D_key, kkc);

    // 4. fused kv outer + value conv + D skip + q gating + exact GELU
    kv_conv_gate_kernel<<<dim3(LL / 128, HH, B_), 256, 0, stream>>>(
        kkc, vb, qb, k_v, Dv, yg);

    // 5. pointwise projection -> d_out (B, 1024, L) f32
    gemm1024_kernel<<<dim3(512, 8), 256, 0, stream>>>(wp_bf, pw_b, yg,
                                                      nullptr, (float*)d_out);
}
